// SRNN2_5377299054670
// MI455X (gfx1250) — matrix operand order repaired and checked by reading, not compiled, not run on hardware
//
#include <hip/hip_runtime.h>
#include <math.h>

// Problem sizes (fixed by the reference)
#define B_    64
#define T_    1000
#define D_    256
#define H_    512
#define O_    20
#define OPAD  32          // out_w cols padded to 2 WMMA n-tiles
#define THR0  0.01f
#define BETA  1.8f

#define NBLK  64          // persistent grid: 64 blocks x 256 threads = 512 waves
#define NTHR  256

typedef __attribute__((ext_vector_type(16))) _Float16 v16h;
typedef __attribute__((ext_vector_type(8)))  float    v8f;

// flat d_out offsets (floats), reference tuple order:
// sum_output, (s1t, s2t, spike_out0), (m1t, m2t, mot)
#define OFF_SUM ((size_t)0)
#define OFF_S1T (OFF_SUM + (size_t)B_*O_)
#define OFF_S2T (OFF_S1T + (size_t)B_*T_*H_)
#define OFF_SPK (OFF_S2T + (size_t)B_*T_*H_)
#define OFF_M1T (OFF_SPK + (size_t)B_*T_*O_)
#define OFF_M2T (OFF_M1T + (size_t)B_*T_*H_)
#define OFF_MOT (OFF_M2T + (size_t)B_*T_*H_)

struct Params {
  // inputs (fp32)
  const float *x, *rec1, *inp1, *bias1, *tau_m1, *tau_adp1;
  const float *rec2, *inp2, *bias2, *tau_m2, *tau_adp2, *out_w, *out_tau;
  // output base
  float* out;
  // f16 packed weights (WMMA B-fragment layout) + dynamic f16 operands
  _Float16 *rec1pk, *inp1pk, *rec2pk, *inp2pk, *outwpk;
  _Float16 *s1h, *s2h, *xth;
  // per-step GEMM results (f32)
  float *cur1, *cur2, *co;
  // recurrent state (f32)
  float *m1, *th1, *s1, *m2, *th2, *s2, *mo, *acc;
  // per-channel decay factors
  float *a1, *r1, *a2, *r2, *ao;
  // grid barrier counter
  unsigned* bar;
};

// ---------------------------------------------------------------------------
// Weight packing: per 32x16 (KxN) tile, store [lane 0..31][8 x u32] contiguous.
// Semantics: u32 j of lane L holds f16 pair (k, k+1) with
//   k = kt*32 + (j<4 ? 0 : 16) + (L<16 ? 0 : 8) + 2(j&3),  n = nt*16 + (L & 15)   (low half = k)
// matching the CDNA5 16-bit B-matrix VGPR layout.
// ---------------------------------------------------------------------------
__device__ inline void pack_pair(const float* W, _Float16* pk,
                                 int Npad, int Nsrc, size_t pi) {
  unsigned j    = (unsigned)(pi & 7u);
  unsigned lane = (unsigned)((pi >> 3) & 31u);
  size_t   rest = pi >> 8;
  unsigned NT   = (unsigned)(Npad >> 4);
  unsigned nt   = (unsigned)(rest % NT);
  size_t   kt   = rest / NT;
  int k = (int)(kt * 32 + ((j >> 2) << 4) + ((lane >> 4) << 3) + 2u * (j & 3u));
  int n = (int)(nt * 16 + (lane & 15u));
  float w0 = (n < Nsrc) ? W[(size_t)k * Nsrc + n]       : 0.f;
  float w1 = (n < Nsrc) ? W[((size_t)k + 1) * Nsrc + n] : 0.f;
  pk[2 * pi]     = (_Float16)w0;
  pk[2 * pi + 1] = (_Float16)w1;
}

// A fragment, row-major f16 source: 16-bit A 16x32 layout.
__device__ inline v16h load_a_frag(const _Float16* A, int lda,
                                   int m0, int k0, int lane) {
  union { v16h h; unsigned u[8]; } r;
  int row = m0 + (lane & 15);
  int kb  = k0 + ((lane >> 4) << 3);            // +8 for lanes 16..31
  const unsigned* pp = (const unsigned*)(A + (size_t)row * lda + kb);
#pragma unroll
  for (int i = 0; i < 4; ++i) r.u[i] = pp[i];       // K = kb+0..7
#pragma unroll
  for (int i = 0; i < 4; ++i) r.u[4 + i] = pp[8 + i]; // K = kb+16..23
  return r.h;
}

// B fragment from packed weights: 8 consecutive dwords per lane.
__device__ inline v16h load_b_frag(const _Float16* pk, int NT,
                                   int kt, int nt, int lane) {
  const unsigned* pp =
      (const unsigned*)pk + (((size_t)kt * NT + nt) * 32 + lane) * 8;
  union { v16h h; unsigned u[8]; } r;
#pragma unroll
  for (int j = 0; j < 8; ++j) r.u[j] = pp[j];
  return r.h;
}

__device__ inline v8f gemm_acc(v8f c, const _Float16* A, int lda,
                               const _Float16* pk, int NT,
                               int mt, int nt, int ktiles, int lane) {
  for (int kt = 0; kt < ktiles; ++kt) {
    if (kt + 1 < ktiles) {
      const unsigned* nx =
          (const unsigned*)pk + (((size_t)(kt + 1) * NT + nt) * 32 + lane) * 8;
      __builtin_prefetch((const void*)nx, 0, 3);   // global_prefetch_b8
    }
    v16h a = load_a_frag(A, lda, mt * 16, kt * 32, lane);
    v16h b = load_b_frag(pk, NT, kt, nt, lane);
    c = __builtin_amdgcn_wmma_f32_16x16x32_f16(
        /*neg_a=*/false, a, /*neg_b=*/false, b,
        /*c_mod=*/(short)0, c, /*reuse_a=*/false, /*reuse_b=*/false);
  }
  return c;
}

// C/D 32-bit layout: VGPR i -> row (i + 8*(lane>=16)), col = lane&15.
__device__ inline void store_c(const v8f& c, float* C, int ldc,
                               int mt, int nt, int lane) {
  int col = nt * 16 + (lane & 15);
  int rb  = mt * 16 + ((lane >> 4) << 3);
#pragma unroll
  for (int i = 0; i < 8; ++i) C[(size_t)(rb + i) * ldc + col] = c[i];
}

// Grid-wide split barrier: arrive via atomic, spin with s_sleep.
__device__ inline void grid_barrier(unsigned* cnt, unsigned bar) {
  __syncthreads();
  if (threadIdx.x == 0) {
    __threadfence();
    atomicAdd(cnt, 1u);
    unsigned tgt = bar * (unsigned)NBLK;
    while (atomicAdd(cnt, 0u) < tgt) __builtin_amdgcn_s_sleep(2);
    __threadfence();
  }
  __syncthreads();
}

// Elementwise adaptive-LIF update (reads/writes only its own element).
__device__ inline void alif(int idx, int h, int b, int t,
                            float* m, float* th, float* s, _Float16* sh,
                            const float* cur, const float* alpha,
                            const float* ro, const float* bias,
                            float* mOut, float* sOut) {
  float al   = alpha[h];
  float r    = ro[h];
  float prev = s[idx];
  float thn  = r * th[idx] + (1.f - r) * prev;
  float Bth  = THR0 + BETA * thn;
  float mm   = m[idx] * al + (1.f - al) * cur[idx] - Bth * prev;  // DT = 1
  float sp   = (mm - Bth + bias[h]) > 0.f ? 1.f : 0.f;
  m[idx] = mm; th[idx] = thn; s[idx] = sp; sh[idx] = (_Float16)sp;
  size_t oo = ((size_t)b * T_ + t) * H_ + h;
  mOut[oo] = mm; sOut[oo] = sp;
}

// ---------------------------------------------------------------------------
// Init kernel: pack weights to f16, precompute decay factors, zero state,
// stage x[:,0,:], zero spike_trace_out, reset barrier counter.
// ---------------------------------------------------------------------------
__global__ void srnn_init(Params p) {
  size_t gid = (size_t)blockIdx.x * blockDim.x + threadIdx.x;
  size_t nth = (size_t)gridDim.x * blockDim.x;

  for (size_t i = gid; i < (size_t)H_ * H_ / 2; i += nth) pack_pair(p.rec1, p.rec1pk, H_, H_, i);
  for (size_t i = gid; i < (size_t)D_ * H_ / 2; i += nth) pack_pair(p.inp1, p.inp1pk, H_, H_, i);
  for (size_t i = gid; i < (size_t)H_ * H_ / 2; i += nth) pack_pair(p.rec2, p.rec2pk, H_, H_, i);
  for (size_t i = gid; i < (size_t)H_ * H_ / 2; i += nth) pack_pair(p.inp2, p.inp2pk, H_, H_, i);
  for (size_t i = gid; i < (size_t)H_ * OPAD / 2; i += nth) pack_pair(p.out_w, p.outwpk, OPAD, O_, i);

  for (size_t i = gid; i < (size_t)B_ * T_ * O_; i += nth) p.out[OFF_SPK + i] = 0.f;

  for (size_t i = gid; i < (size_t)B_ * H_; i += nth) {
    p.m1[i] = 0.f; p.th1[i] = THR0; p.s1[i] = 0.f; p.s1h[i] = (_Float16)0.f;
    p.m2[i] = 0.f; p.th2[i] = THR0; p.s2[i] = 0.f; p.s2h[i] = (_Float16)0.f;
  }
  for (size_t i = gid; i < (size_t)B_ * O_; i += nth) { p.mo[i] = 0.f; p.acc[i] = 0.f; }

  for (size_t i = gid; i < (size_t)H_; i += nth) {
    p.a1[i] = expf(-1.f / p.tau_m1[i]);
    p.r1[i] = expf(-1.f / p.tau_adp1[i]);
    p.a2[i] = expf(-1.f / p.tau_m2[i]);
    p.r2[i] = expf(-1.f / p.tau_adp2[i]);
  }
  for (size_t i = gid; i < (size_t)O_; i += nth) p.ao[i] = expf(-1.f / p.out_tau[i]);

  for (size_t i = gid; i < (size_t)B_ * D_; i += nth) {
    size_t b = i >> 8, d = i & (D_ - 1);
    p.xth[i] = (_Float16)p.x[(b * T_ + 0) * D_ + d];   // x[:,0,:]
  }
  if (gid == 0) *p.bar = 0u;
}

// ---------------------------------------------------------------------------
// Persistent scan kernel: for each timestep, Phase A (3 WMMA GEMMs on the
// previous-step spikes), grid barrier, Phase B (elementwise ALIF + softmax
// accumulation + next-x f16 staging), grid barrier.
// ---------------------------------------------------------------------------
__global__ __launch_bounds__(NTHR) void srnn_run(Params p) {
  const int lane = threadIdx.x & 31;
  const int gw   = blockIdx.x * (NTHR / 32) + (threadIdx.x >> 5);
  const int NW   = NBLK * (NTHR / 32);
  const int gtid = blockIdx.x * NTHR + threadIdx.x;
  const int NTH  = NBLK * NTHR;
  unsigned bar = 0;

  for (int t = 0; t < T_; ++t) {
    // ---- Phase A: 128 + 128 + 8 = 264 output tiles of 16x16 ----
    for (int tile = gw; tile < 264; tile += NW) {
      v8f c = {0.f, 0.f, 0.f, 0.f, 0.f, 0.f, 0.f, 0.f};
      if (tile < 128) {                       // cur1 = s1@rec1 + x_t@inp1
        int mt = tile >> 5, nt = tile & 31;
        c = gemm_acc(c, p.s1h, H_, p.rec1pk, H_ / 16, mt, nt, H_ / 32, lane);
        c = gemm_acc(c, p.xth, D_, p.inp1pk, H_ / 16, mt, nt, D_ / 32, lane);
        store_c(c, p.cur1, H_, mt, nt, lane);
      } else if (tile < 256) {                // cur2 = s2@rec2 + s1@inp2
        int q = tile - 128; int mt = q >> 5, nt = q & 31;
        c = gemm_acc(c, p.s2h, H_, p.rec2pk, H_ / 16, mt, nt, H_ / 32, lane);
        c = gemm_acc(c, p.s1h, H_, p.inp2pk, H_ / 16, mt, nt, H_ / 32, lane);
        store_c(c, p.cur2, H_, mt, nt, lane);
      } else {                                // co = s2@out_w  (N padded to 32)
        int q = tile - 256; int mt = q >> 1, nt = q & 1;
        c = gemm_acc(c, p.s2h, H_, p.outwpk, OPAD / 16, mt, nt, H_ / 32, lane);
        store_c(c, p.co, OPAD, mt, nt, lane);
      }
    }
    grid_barrier(p.bar, ++bar);

    // ---- Phase B: elementwise state updates + trace writes ----
    for (int e = gtid; e < B_ * H_; e += NTH) {
      int h = e & (H_ - 1), b = e >> 9;
      alif(e, h, b, t, p.m1, p.th1, p.s1, p.s1h, p.cur1, p.a1, p.r1, p.bias1,
           p.out + OFF_M1T, p.out + OFF_S1T);
    }
    for (int e = gtid; e < B_ * H_; e += NTH) {
      int h = e & (H_ - 1), b = e >> 9;
      alif(e, h, b, t, p.m2, p.th2, p.s2, p.s2h, p.cur2, p.a2, p.r2, p.bias2,
           p.out + OFF_M2T, p.out + OFF_S2T);
    }
    if (gtid < B_) {                          // output leak + softmax accum
      int b = gtid;
      float mrow[O_], er[O_];
      float mx = -3.4e38f;
      for (int o = 0; o < O_; ++o) {
        float aoo = p.ao[o];
        float mn  = p.mo[b * O_ + o] * aoo + (1.f - aoo) * p.co[b * OPAD + o];
        p.mo[b * O_ + o] = mn;
        p.out[OFF_MOT + ((size_t)b * T_ + t) * O_ + o] = mn;
        mrow[o] = mn;
        mx = fmaxf(mx, mn);
      }
      float sum = 0.f;
      for (int o = 0; o < O_; ++o) { er[o] = expf(mrow[o] - mx); sum += er[o]; }
      float inv = 1.f / sum;
      for (int o = 0; o < O_; ++o) p.acc[b * O_ + o] += er[o] * inv;
    }
    if (t + 1 < T_) {                         // stage x[:,t+1,:] as f16
      for (int e = gtid; e < B_ * D_; e += NTH) {
        int d = e & (D_ - 1), b = e >> 8;
        p.xth[e] = (_Float16)p.x[((size_t)b * T_ + (t + 1)) * D_ + d];
      }
    }
    grid_barrier(p.bar, ++bar);
  }

  // sum_output = acc
  for (int e = gtid; e < B_ * O_; e += NTH) p.out[e] = p.acc[e];
}

// ---------------------------------------------------------------------------
extern "C" void kernel_launch(void* const* d_in, const int* in_sizes, int n_in,
                              void* d_out, int out_size, void* d_ws, size_t ws_size,
                              hipStream_t stream) {
  (void)in_sizes; (void)n_in; (void)out_size; (void)ws_size;
  Params p;
  p.x       = (const float*)d_in[0];
  p.rec1    = (const float*)d_in[1];
  p.inp1    = (const float*)d_in[2];
  p.bias1   = (const float*)d_in[3];
  p.tau_m1  = (const float*)d_in[4];
  p.tau_adp1= (const float*)d_in[5];
  p.rec2    = (const float*)d_in[6];
  p.inp2    = (const float*)d_in[7];
  p.bias2   = (const float*)d_in[8];
  p.tau_m2  = (const float*)d_in[9];
  p.tau_adp2= (const float*)d_in[10];
  p.out_w   = (const float*)d_in[11];
  p.out_tau = (const float*)d_in[12];
  p.out     = (float*)d_out;

  char* w = (char*)d_ws;
  size_t off = 0;
  auto carve = [&](size_t bytes) -> char* {
    char* r = w + off;
    off = (off + bytes + 255) & ~(size_t)255;
    return r;
  };
  p.rec1pk = (_Float16*)carve((size_t)H_ * H_ * 2);
  p.inp1pk = (_Float16*)carve((size_t)D_ * H_ * 2);
  p.rec2pk = (_Float16*)carve((size_t)H_ * H_ * 2);
  p.inp2pk = (_Float16*)carve((size_t)H_ * H_ * 2);
  p.outwpk = (_Float16*)carve((size_t)H_ * OPAD * 2);
  p.s1h    = (_Float16*)carve((size_t)B_ * H_ * 2);
  p.s2h    = (_Float16*)carve((size_t)B_ * H_ * 2);
  p.xth    = (_Float16*)carve((size_t)B_ * D_ * 2);
  p.cur1   = (float*)carve((size_t)B_ * H_ * 4);
  p.cur2   = (float*)carve((size_t)B_ * H_ * 4);
  p.co     = (float*)carve((size_t)B_ * OPAD * 4);
  p.m1     = (float*)carve((size_t)B_ * H_ * 4);
  p.th1    = (float*)carve((size_t)B_ * H_ * 4);
  p.s1     = (float*)carve((size_t)B_ * H_ * 4);
  p.m2     = (float*)carve((size_t)B_ * H_ * 4);
  p.th2    = (float*)carve((size_t)B_ * H_ * 4);
  p.s2     = (float*)carve((size_t)B_ * H_ * 4);
  p.mo     = (float*)carve((size_t)B_ * O_ * 4);
  p.acc    = (float*)carve((size_t)B_ * O_ * 4);
  p.a1     = (float*)carve((size_t)H_ * 4);
  p.r1     = (float*)carve((size_t)H_ * 4);
  p.a2     = (float*)carve((size_t)H_ * 4);
  p.r2     = (float*)carve((size_t)H_ * 4);
  p.ao     = (float*)carve((size_t)O_ * 4);
  p.bar    = (unsigned*)carve(256);

  srnn_init<<<512, 256, 0, stream>>>(p);
  srnn_run<<<NBLK, NTHR, 0, stream>>>(p);
}
